// NISER_ODE_63548336112064
// MI455X (gfx1250) — matrix operand order repaired and checked by reading, not compiled, not run on hardware
//
#include <hip/hip_runtime.h>

typedef __attribute__((ext_vector_type(16))) _Float16 v16h;
typedef __attribute__((ext_vector_type(8)))  float    v8f;
typedef __attribute__((ext_vector_type(4)))  int      v4i;

typedef __attribute__((address_space(1))) v4i gv4i;
typedef __attribute__((address_space(3))) v4i lv4i;

#if __has_builtin(__builtin_amdgcn_global_load_async_to_lds_b128) && \
    __has_builtin(__builtin_amdgcn_s_wait_asynccnt)
#define USE_ASYNC 1
#else
#define USE_ASYNC 0
#endif

#define V_ 100000
#define D_ 128
#define B_ 512
#define L_ 20
#define T_ 20
#define H_ 32
#define C_ 33
#define N_ (B_*L_)
#define SCALE_ 12.0f

static __device__ __forceinline__ float warpSum(float v) {
#pragma unroll
  for (int off = 16; off > 0; off >>= 1) v += __shfl_xor(v, off, 32);
  return v;
}

// A-matrix (16-bit, 16x32) lane packing: element e -> K = koff + e,
// koff = kstep*32 + (lane>>4)*16
static __device__ __forceinline__ v16h load_a(const _Float16* row, int koff) {
  v16h r;
#pragma unroll
  for (int e = 0; e < 16; ++e) r[e] = row[koff + e];
  return r;
}

// B-matrix (16-bit, 32x16) lane packing: lane = column, 16 contiguous K halves,
// kbase = kstep*32 + (lane>>4)*16
static __device__ __forceinline__ v16h load_b(const _Float16* row, int kbase) {
  v16h r;
#pragma unroll
  for (int e = 0; e < 16; ++e) r[e] = row[kbase + e];
  return r;
}

// ---------------- K1: featn = emb[iid]/||.||  (10240 rows) ----------------
__global__ __launch_bounds__(256) void feat_kernel(const float* __restrict__ emb,
                                                   const int* __restrict__ iid,
                                                   float* __restrict__ featn) {
  int row = blockIdx.x * 8 + (threadIdx.x >> 5);
  int lane = threadIdx.x & 31;
  if (row >= N_) return;
  const float4* src = (const float4*)(emb + (size_t)iid[row] * D_);
  float4 x = src[lane];
  float ss = warpSum(x.x*x.x + x.y*x.y + x.z*x.z + x.w*x.w);
  float inv = rsqrtf(fmaxf(ss, 1e-24f));
  float4 o; o.x = x.x*inv; o.y = x.y*inv; o.z = x.z*inv; o.w = x.w*inv;
  ((float4*)(featn + (size_t)row * D_))[lane] = o;
}

// ---------------- K2: embn_h = f16(l2n(emb))  (100000 rows) ----------------
__global__ __launch_bounds__(256) void embn_kernel(const float* __restrict__ emb,
                                                   _Float16* __restrict__ embh) {
  int row = blockIdx.x * 8 + (threadIdx.x >> 5);
  int lane = threadIdx.x & 31;
  if (row >= V_) return;
  const float4* src = (const float4*)(emb + (size_t)row * D_);
  float4 x = src[lane];
  float ss = warpSum(x.x*x.x + x.y*x.y + x.z*x.z + x.w*x.w);
  float inv = 1.0f / (sqrtf(ss) + 1e-12f);
  _Float16* dst = embh + (size_t)row * D_ + lane * 4;
  dst[0] = (_Float16)(x.x*inv); dst[1] = (_Float16)(x.y*inv);
  dst[2] = (_Float16)(x.z*inv); dst[3] = (_Float16)(x.w*inv);
}

// ---------------- K3: f32 -> f16 cast (cde_w2) ----------------
__global__ void h16_kernel(const float* __restrict__ src, _Float16* __restrict__ dst, int n) {
  int i = blockIdx.x * blockDim.x + threadIdx.x;
  if (i < n) dst[i] = (_Float16)src[i];
}

// ---------------- K4: neural CDE, 16 sessions/block, WMMA layer-2 ----------------
__global__ __launch_bounds__(256) void cde_kernel(
    const float* __restrict__ emb, const float* __restrict__ red_w,
    const float* __restrict__ red_b, const float* __restrict__ cde_w1,
    const float* __restrict__ cde_b1, const _Float16* __restrict__ w2h,
    const float* __restrict__ cde_b2, const float* __restrict__ init_w,
    const float* __restrict__ init_b, const float* __restrict__ rec_w,
    const float* __restrict__ rec_b, const float* __restrict__ times,
    const int* __restrict__ embeds_ids, float* __restrict__ te_out) {
  __shared__ float    Xs[16][T_][C_];
  __shared__ float    zcur[16][H_];
  __shared__ float    zst[16][H_];
  __shared__ float    kprev[16][H_];
  __shared__ float    kcur[16][H_];
  __shared__ float    gacc[16][H_];
  __shared__ float    dxs[16][C_];
  __shared__ _Float16 h1h[16][D_];

  const int tid = threadIdx.x;
  const int m0 = blockIdx.x * 16;

  // fode raw: y = emb[embeds_ids] @ red_w.T + red_b
  for (int i = tid; i < 16 * T_ * H_; i += 256) {
    int m = i / (T_ * H_), r = i % (T_ * H_), t = r / H_, h = r % H_;
    const float* er = emb + (size_t)embeds_ids[(m0 + m) * T_ + t] * D_;
    const float* wr = red_w + h * D_;
    float acc = red_b[h];
    for (int k = 0; k < D_; ++k) acc += er[k] * wr[k];
    Xs[m][t][1 + h] = acc;
  }
  __syncthreads();
  for (int i = tid; i < 16 * T_; i += 256) {
    int m = i / T_, t = i % T_;
    float ss = 0.f;
    for (int h = 0; h < H_; ++h) { float v = Xs[m][t][1 + h]; ss += v * v; }
    float inv = 1.0f / (sqrtf(ss) + 1e-12f);
    for (int h = 0; h < H_; ++h) Xs[m][t][1 + h] *= inv;
    Xs[m][t][0] = times[(m0 + m) * T_ + t];
  }
  __syncthreads();
  for (int i = tid; i < 16 * H_; i += 256) {
    int m = i / H_, h = i % H_;
    const float* wr = init_w + h * C_;
    float acc = init_b[h];
    for (int c = 0; c < C_; ++c) acc += Xs[m][0][c] * wr[c];
    zcur[m][h] = acc;
  }
  __syncthreads();

  const int wave = tid >> 5, lane = tid & 31;
  const int arow = lane & 15;
  const int khalf = lane >> 4;

  for (int t = 0; t < T_ - 1; ++t) {
    for (int i = tid; i < 16 * C_; i += 256) {
      int m = i / C_, c = i % C_;
      dxs[m][c] = Xs[m][t + 1][c] - Xs[m][t][c];
    }
    for (int i = tid; i < 16 * H_; i += 256) gacc[i / H_][i % H_] = 0.f;
    __syncthreads();

#pragma unroll 1
    for (int s = 0; s < 4; ++s) {
      const float cin = (s == 1 || s == 2) ? 0.5f : 1.0f;
      const float wrk = (s == 1 || s == 2) ? 2.0f : 1.0f;
      for (int i = tid; i < 16 * H_; i += 256) {
        int m = i / H_, h = i % H_;
        zst[m][h] = (s == 0) ? zcur[m][h] : zcur[m][h] + cin * kprev[m][h];
        kcur[m][h] = 0.f;
      }
      __syncthreads();
      for (int i = tid; i < 16 * D_; i += 256) {
        int m = i >> 7, j = i & 127;
        const float* w1r = cde_w1 + j * H_;
        float acc = cde_b1[j];
        for (int h = 0; h < H_; ++h) acc += zst[m][h] * w1r[h];
        h1h[m][j] = (_Float16)fmaxf(acc, 0.f);
      }
      __syncthreads();
      v16h a0 = load_a(&h1h[arow][0], 0 * 32 + khalf * 16);
      v16h a1 = load_a(&h1h[arow][0], 1 * 32 + khalf * 16);
      v16h a2 = load_a(&h1h[arow][0], 2 * 32 + khalf * 16);
      v16h a3 = load_a(&h1h[arow][0], 3 * 32 + khalf * 16);
      for (int nt = wave; nt < 66; nt += 8) {
        const _Float16* brow = w2h + (size_t)(nt * 16 + arow) * D_;
        if (nt + 8 < 66)
          __builtin_prefetch(w2h + (size_t)((nt + 8) * 16 + arow) * D_, 0, 1);
        v8f c = {};
        c = __builtin_amdgcn_wmma_f32_16x16x32_f16(false, a0, false, load_b(brow, 0 * 32 + khalf * 16), (short)0, c, false, false);
        c = __builtin_amdgcn_wmma_f32_16x16x32_f16(false, a1, false, load_b(brow, 1 * 32 + khalf * 16), (short)0, c, false, false);
        c = __builtin_amdgcn_wmma_f32_16x16x32_f16(false, a2, false, load_b(brow, 2 * 32 + khalf * 16), (short)0, c, false, false);
        c = __builtin_amdgcn_wmma_f32_16x16x32_f16(false, a3, false, load_b(brow, 3 * 32 + khalf * 16), (short)0, c, false, false);
        const int n = nt * 16 + arow;
        const int hh = n / C_, cc = n % C_;
        const int mb = (lane < 16) ? 0 : 8;
        const float b2v = cde_b2[n];
#pragma unroll
        for (int r = 0; r < 8; ++r) {
          float val = tanhf(c[r] + b2v);
          atomicAdd(&kcur[mb + r][hh], val * dxs[mb + r][cc]);
        }
      }
      __syncthreads();
      for (int i = tid; i < 16 * H_; i += 256) {
        int m = i / H_, h = i % H_;
        float k = kcur[m][h];
        gacc[m][h] += wrk * k;
        kprev[m][h] = k;
      }
      __syncthreads();
    }
    for (int i = tid; i < 16 * H_; i += 256) {
      int m = i / H_, h = i % H_;
      zcur[m][h] += gacc[m][h] * (1.0f / 6.0f);
    }
    __syncthreads();
  }
  for (int i = tid; i < 16 * D_; i += 256) {
    int m = i >> 7, j = i & 127;
    const float* wr = rec_w + j * H_;
    float acc = rec_b[j];
    for (int h = 0; h < H_; ++h) acc += zcur[m][h] * wr[h];
    te_out[(size_t)(m0 + m) * D_ + j] = acc;
  }
}

// ---------------- K5: attention readout + sr (f16) ----------------
__global__ __launch_bounds__(128) void att_kernel(
    const float* __restrict__ featn, const float* __restrict__ fcu_w,
    const float* __restrict__ fcv_w, const float* __restrict__ fcv_b,
    const float* __restrict__ fce_w, const float* __restrict__ fcsr_w,
    const float* __restrict__ te, const int* __restrict__ last_nodes,
    _Float16* __restrict__ srh) {
  __shared__ float fns[L_][D_];
  __shared__ float lastrow[D_];
  __shared__ float fvv[D_];
  __shared__ float ea[L_];
  __shared__ float al[L_];
  __shared__ float srg[D_];
  __shared__ float red[4];
  const int b = blockIdx.x;
  const int j = threadIdx.x;
  const int lane = j & 31, wv = j >> 5;

  for (int i = j; i < L_ * D_; i += 128)
    fns[i / D_][i % D_] = featn[(size_t)(b * L_ + i / D_) * D_ + (i % D_)];
  lastrow[j] = featn[(size_t)last_nodes[b] * D_ + j];
  __syncthreads();

  {
    const float* wr = fcv_w + j * D_;
    float acc = fcv_b[j];
    for (int k = 0; k < D_; ++k) acc += lastrow[k] * wr[k];
    fvv[j] = acc;
  }
  __syncthreads();

  for (int i0 = 0; i0 < L_; ++i0) {
    const float* wr = fcu_w + j * D_;
    float fu = 0.f;
    for (int k = 0; k < D_; ++k) fu += fns[i0][k] * wr[k];
    float sg = 1.0f / (1.0f + __expf(-(fu + fvv[j])));
    float contrib = warpSum(sg * fce_w[j]);
    if (lane == 0) red[wv] = contrib;
    __syncthreads();
    if (j == 0) ea[i0] = red[0] + red[1] + red[2] + red[3];
    __syncthreads();
  }
  if (j == 0) {
    float mx = -1e30f;
    for (int i = 0; i < L_; ++i) mx = fmaxf(mx, ea[i]);
    float s = 0.f;
    for (int i = 0; i < L_; ++i) { al[i] = __expf(ea[i] - mx); s += al[i]; }
    for (int i = 0; i < L_; ++i) al[i] /= s;
  }
  __syncthreads();
  {
    float acc = 0.f;
    for (int i = 0; i < L_; ++i) acc += al[i] * fns[i][j];
    srg[j] = acc;
  }
  __syncthreads();
  float v = te[(size_t)b * D_ + j];
  const float* wr = fcsr_w + j * (2 * D_);
  for (int k = 0; k < D_; ++k) v += lastrow[k] * wr[k];
  for (int k = 0; k < D_; ++k) v += srg[k] * wr[D_ + k];
  float ss = warpSum(v * v);
  if (lane == 0) red[wv] = ss;
  __syncthreads();
  float inv = 1.0f / (sqrtf(red[0] + red[1] + red[2] + red[3]) + 1e-12f);
  srh[(size_t)b * D_ + j] = (_Float16)(v * inv);
}

// ---------------- K6: logits tiles (WMMA); pass1 -> rowsum of exp(v-12),
//                  pass2 -> out = v - rowlse.  B tile staged in LDS (async). ----
__global__ __launch_bounds__(256) void logits_pass_kernel(
    const _Float16* __restrict__ srh, const _Float16* __restrict__ embh,
    float* __restrict__ rowsum, const float* __restrict__ rowlse,
    float* __restrict__ out, int final_pass) {
  __shared__ _Float16 Bs[16][D_];
  const int ntile = blockIdx.x;
  {
    int rr = threadIdx.x >> 4, ck = threadIdx.x & 15;
    const _Float16* g = embh + (size_t)(ntile * 16 + rr) * D_ + ck * 8;
    _Float16* l = &Bs[rr][ck * 8];
#if USE_ASYNC
    __builtin_amdgcn_global_load_async_to_lds_b128((gv4i*)g, (lv4i*)l, 0, 0);
#else
    *(float4*)l = *(const float4*)g;
#endif
  }
#if USE_ASYNC
  __builtin_amdgcn_s_wait_asynccnt(0);
#endif
  __syncthreads();

  const int lane = threadIdx.x & 31, wave = threadIdx.x >> 5;
  const int arow = lane & 15, khalf = lane >> 4;
  const v16h b0 = load_b(&Bs[arow][0], 0 * 32 + khalf * 16);
  const v16h b1 = load_b(&Bs[arow][0], 1 * 32 + khalf * 16);
  const v16h b2 = load_b(&Bs[arow][0], 2 * 32 + khalf * 16);
  const v16h b3 = load_b(&Bs[arow][0], 3 * 32 + khalf * 16);
  const int n = ntile * 16 + arow;

#pragma unroll
  for (int mi = 0; mi < 4; ++mi) {
    const int mtile = wave + mi * 8;
    const _Float16* ap = srh + (size_t)(mtile * 16 + arow) * D_;
    v8f c = {};
    c = __builtin_amdgcn_wmma_f32_16x16x32_f16(false, load_a(ap, 0 * 32 + khalf * 16), false, b0, (short)0, c, false, false);
    c = __builtin_amdgcn_wmma_f32_16x16x32_f16(false, load_a(ap, 1 * 32 + khalf * 16), false, b1, (short)0, c, false, false);
    c = __builtin_amdgcn_wmma_f32_16x16x32_f16(false, load_a(ap, 2 * 32 + khalf * 16), false, b2, (short)0, c, false, false);
    c = __builtin_amdgcn_wmma_f32_16x16x32_f16(false, load_a(ap, 3 * 32 + khalf * 16), false, b3, (short)0, c, false, false);
    const int mb = mtile * 16 + ((lane < 16) ? 0 : 8);
    if (!final_pass) {
#pragma unroll
      for (int r = 0; r < 8; ++r) {
        // fixed-reference sumexp: SCALE*logit <= 12 (both operands l2-normalized)
        float ex = __expf(SCALE_ * c[r] - 12.0f);
#pragma unroll
        for (int off = 1; off < 16; off <<= 1) ex += __shfl_xor(ex, off, 32);
        if (arow == 0) atomicAdd(&rowsum[mb + r], ex);
      }
    } else {
#pragma unroll
      for (int r = 0; r < 8; ++r)
        out[(size_t)(mb + r) * V_ + n] = SCALE_ * c[r] - rowlse[mb + r];
    }
  }
}

// ---------------- K7: rowsum init / lse ----------------
__global__ void zero_kernel(float* __restrict__ p, int n) {
  int i = blockIdx.x * blockDim.x + threadIdx.x;
  if (i < n) p[i] = 0.f;
}
__global__ void lse_kernel(const float* __restrict__ rowsum, float* __restrict__ rowlse) {
  int i = blockIdx.x * blockDim.x + threadIdx.x;
  if (i < B_) rowlse[i] = 12.0f + __logf(rowsum[i]);
}

extern "C" void kernel_launch(void* const* d_in, const int* in_sizes, int n_in,
                              void* d_out, int out_size, void* d_ws, size_t ws_size,
                              hipStream_t stream) {
  const float* emb    = (const float*)d_in[0];
  const float* fcu_w  = (const float*)d_in[7];
  const float* fcv_w  = (const float*)d_in[8];
  const float* fcv_b  = (const float*)d_in[9];
  const float* fce_w  = (const float*)d_in[10];
  const float* fcsr_w = (const float*)d_in[11];
  const float* red_w  = (const float*)d_in[12];
  const float* red_b  = (const float*)d_in[13];
  const float* rec_w  = (const float*)d_in[14];
  const float* rec_b  = (const float*)d_in[15];
  const float* cde_w1 = (const float*)d_in[16];
  const float* cde_b1 = (const float*)d_in[17];
  const float* cde_w2 = (const float*)d_in[18];
  const float* cde_b2 = (const float*)d_in[19];
  const float* init_w = (const float*)d_in[20];
  const float* init_b = (const float*)d_in[21];
  const float* times  = (const float*)d_in[23];
  const int*   iid    = (const int*)d_in[24];
  const int*   last_nodes = (const int*)d_in[28];
  const int*   embeds_ids = (const int*)d_in[29];
  float* out = (float*)d_out;

  char* ws = (char*)d_ws;
  float*    featn  = (float*)(ws + 0);                      // 10240*128 f32
  _Float16* embh   = (_Float16*)(ws + 5242880);             // 100000*128 f16
  _Float16* w2h    = (_Float16*)(ws + 30842880);            // 1056*128 f16
  float*    te     = (float*)(ws + 31113216);               // 512*128 f32
  _Float16* srh    = (_Float16*)(ws + 31375360);             // 512*128 f16
  float*    rowsum = (float*)(ws + 31506432);               // 512 f32
  float*    rowlse = (float*)(ws + 31508480);               // 512 f32

  feat_kernel<<<N_ / 8, 256, 0, stream>>>(emb, iid, featn);
  embn_kernel<<<V_ / 8, 256, 0, stream>>>(emb, embh);
  h16_kernel<<<(1056 * 128 + 255) / 256, 256, 0, stream>>>(cde_w2, w2h, 1056 * 128);
  zero_kernel<<<2, 256, 0, stream>>>(rowsum, B_);
  cde_kernel<<<B_ / 16, 256, 0, stream>>>(emb, red_w, red_b, cde_w1, cde_b1, w2h,
                                          cde_b2, init_w, init_b, rec_w, rec_b,
                                          times, embeds_ids, te);
  att_kernel<<<B_, 128, 0, stream>>>(featn, fcu_w, fcv_w, fcv_b, fce_w, fcsr_w,
                                     te, last_nodes, srh);
  logits_pass_kernel<<<V_ / 16, 256, 0, stream>>>(srh, embh, rowsum, rowlse, out, 0);
  lse_kernel<<<2, 256, 0, stream>>>(rowsum, rowlse);
  logits_pass_kernel<<<V_ / 16, 256, 0, stream>>>(srh, embh, rowsum, rowlse, out, 1);
}
